// SwinTransformerBlock3D_15728170238470
// MI455X (gfx1250) — compile-verified, matrix operand order checked by reading, not run on hardware
//
#include <hip/hip_runtime.h>
#include <hip/hip_bf16.h>
#include <math.h>

typedef __bf16 bf16_t;
typedef bf16_t bf16x16 __attribute__((ext_vector_type(16)));
typedef bf16_t bf16x8  __attribute__((ext_vector_type(8)));
typedef float  f32x8   __attribute__((ext_vector_type(8)));
typedef float  f32x4   __attribute__((ext_vector_type(4)));

#define WMMA_BF16(a, b, c) \
  __builtin_amdgcn_wmma_f32_16x16x32_bf16(false, (a), false, (b), (short)0, (c), false, false)

// LDS row strides (elements), padded to break 16-row bank-conflict patterns
// while keeping every fragment base 16-byte aligned.
#define XN_LD  104
#define QK_LD  200
#define VT_LD  136
#define SC_LD  132
#define AO_LD  104
#define HID_LD 392

__device__ inline float wsum(float v){
  #pragma unroll
  for (int o = 16; o > 0; o >>= 1) v += __shfl_xor(v, o, 32);
  return v;
}
__device__ inline float wmaxr(float v){
  #pragma unroll
  for (int o = 16; o > 0; o >>= 1) v = fmaxf(v, __shfl_xor(v, o, 32));
  return v;
}

// A fragment (16x32 bf16, row-major source): two contiguous 128-bit loads/lane.
// Per ISA 7.12.2: lane<16 holds row M=lane, K=kb..kb+7 then 16+kb..16+kb+7 (kb = (lane>>4)*8).
// Also serves as B-fragment loader for a TRANSPOSED source (B[k][n] = src[n][k]).
__device__ inline bf16x16 ldA(const bf16_t* __restrict__ p0, int ld, int lane){
  int m = lane & 15, kb = (lane >> 4) << 3;
  const bf16_t* p = p0 + m * ld + kb;
  bf16x8 lo = *(const bf16x8*)(p);
  bf16x8 hi = *(const bf16x8*)(p + 16);
  return __builtin_shufflevector(lo, hi, 0,1,2,3,4,5,6,7,8,9,10,11,12,13,14,15);
}
// Same A layout converting from f32 source (softmax probabilities): 4x float4 loads.
__device__ inline bf16x16 ldAf(const float* __restrict__ p0, int ld, int lane){
  int m = lane & 15, kb = (lane >> 4) << 3;
  const float* p = p0 + m * ld + kb;
  f32x4 a0 = *(const f32x4*)(p);
  f32x4 a1 = *(const f32x4*)(p + 4);
  f32x4 a2 = *(const f32x4*)(p + 16);
  f32x4 a3 = *(const f32x4*)(p + 20);
  bf16x16 a;
  #pragma unroll
  for (int i = 0; i < 4; ++i){
    a[i]      = (bf16_t)a0[i];
    a[4 + i]  = (bf16_t)a1[i];
    a[8 + i]  = (bf16_t)a2[i];
    a[12 + i] = (bf16_t)a3[i];
  }
  return a;
}
// B fragment from pre-swizzled weights: one contiguous 32-byte load per lane.
__device__ inline bf16x16 ldBsw(const bf16_t* __restrict__ w, int tile, int lane){
  return *(const bf16x16*)(w + ((size_t)tile * 32 + lane) * 16);
}

// Convert f32 weights (K x N row-major) into fragment-swizzled bf16:
// dst[((kt*NT+nt)*32 + lane)*16 + e] = W[kt*32 + krel(lane,e)][nt*16 + (lane&15)]
__global__ void swz_bf16(const float* __restrict__ s, bf16_t* __restrict__ d,
                         int N, int NT, int n){
  int i = blockIdx.x * 256 + threadIdx.x;
  if (i >= n) return;
  int t    = i >> 9;          // tile
  int r    = i & 511;
  int lane = r >> 4;
  int e    = r & 15;
  int nt   = t % NT, kt = t / NT;
  int kb   = (lane >> 4) << 3;
  int krel = (e < 8) ? (kb + e) : (16 + kb + (e - 8));
  int k    = kt * 32 + krel;
  int col  = nt * 16 + (lane & 15);
  d[i] = (bf16_t)s[(size_t)k * N + col];
}

// ---------------------------------------------------------------------------
// Kernel A: one workgroup per shifted window (4096 windows, 8 wave32s).
// LN1 -> QKV (v stored transposed) -> per-head scores+bias+mask, softmax,
// probs@v -> proj -> first residual scattered via window-reverse/unshift.
// Dynamic LDS: 221600 B (CDNA5 WGP has 320 KB).
// ---------------------------------------------------------------------------
__global__ __launch_bounds__(256) void attn_kernel(
    const float* __restrict__ x, const float* __restrict__ mask,
    const float* __restrict__ n1w, const float* __restrict__ n1b,
    const float* __restrict__ qkvb, const float* __restrict__ rpb,
    const float* __restrict__ projb,
    const bf16_t* __restrict__ wqkv, const bf16_t* __restrict__ wproj,
    float* __restrict__ xres)
{
  extern __shared__ char smem[];
  int*    tokpos = (int*)smem;                 //   448
  bf16_t* xn   = (bf16_t*)(smem + 448);        // 112*104*2 = 23296
  bf16_t* qk   = (bf16_t*)(smem + 23744);      // 112*200*2 = 44800
  bf16_t* vT   = (bf16_t*)(smem + 68544);      //  96*136*2 = 26112
  float*  sc   = (float*)(smem + 94656);       // 112*132*4 = 59136
  bf16_t* ao   = (bf16_t*)(smem + 153792);     // 112*104*2 = 23296
  float*  rpbs = (float*)(smem + 177088);      // 507*3*4   =  6084
  float*  msk  = (float*)(smem + 183184);      // 98*98*4   = 38416 -> 221600
  const int lane = threadIdx.x & 31;
  const int wv   = threadIdx.x >> 5;
  const int w    = blockIdx.x;
  const int bb   = w >> 11;        // batch
  const int widx = w & 2047;       // window index within batch (mask row)
  const int wid  = widx >> 8;
  const int wih  = (widx >> 4) & 15;
  const int wiw  = widx & 15;

  // ---- stage mask + rel-pos table into LDS (read 3x per window otherwise) ----
  {
    const f32x4* ms = (const f32x4*)(mask + (size_t)widx * 9604);
    f32x4* md = (f32x4*)msk;
    for (int i = threadIdx.x; i < 2401; i += 256) md[i] = ms[i];
    for (int i = threadIdx.x; i < 1521; i += 256) rpbs[i] = rpb[i];
  }

  // ---- gather shifted window + LN1 -> xn (bf16), wave per token ----
  for (int it = 0; it < 13; ++it){
    int t = wv + it * 8;
    if (t < 98){
      int td = t / 49, rr = t - td * 49, th = rr / 7, tw = rr - th * 7;
      int sd = (wid * 2 + td + 1) & 15;          // roll(-1) on D=16
      int sh = (wih * 7 + th + 3) % 112;         // roll(-3) on H=112
      int sw = (wiw * 7 + tw + 3) % 112;         // roll(-3) on W=112
      int gp = ((bb * 16 + sd) * 112 + sh) * 112 + sw;
      if (lane == 0) tokpos[t] = gp;
      const float* xp = x + (size_t)gp * 96;
      float v0 = xp[lane], v1 = xp[lane + 32], v2 = xp[lane + 64];
      float s  = wsum(v0 + v1 + v2);
      float sq = wsum(v0*v0 + v1*v1 + v2*v2);
      float mean = s * (1.f / 96.f);
      float rstd = rsqrtf(sq * (1.f / 96.f) - mean * mean + 1e-5f);
      xn[t*XN_LD + lane     ] = (bf16_t)((v0 - mean) * rstd * n1w[lane     ] + n1b[lane     ]);
      xn[t*XN_LD + lane + 32] = (bf16_t)((v1 - mean) * rstd * n1w[lane + 32] + n1b[lane + 32]);
      xn[t*XN_LD + lane + 64] = (bf16_t)((v2 - mean) * rstd * n1w[lane + 64] + n1b[lane + 64]);
    }
  }
  for (int i = threadIdx.x; i < 14 * XN_LD; i += 256) xn[98 * XN_LD + i] = (bf16_t)0.f;
  for (int i = threadIdx.x; i < 96 * 16;   i += 256){
    int hd = i >> 4, c = 112 + (i & 15);
    vT[hd * VT_LD + c] = (bf16_t)0.f;          // zero padded key tokens in v^T
  }
  __syncthreads();

  // ---- QKV GEMM: (112x96)@(96x288); q pre-scaled; v written transposed ----
  for (int t = wv; t < 126; t += 8){
    int mt = t / 18, nt = t - mt * 18;
    int m0 = mt * 16, n0 = nt * 16;
    f32x8 acc = {};
    #pragma unroll
    for (int kk = 0; kk < 96; kk += 32){
      bf16x16 a = ldA(xn + m0 * XN_LD + kk, XN_LD, lane);
      bf16x16 b = ldBsw(wqkv, (kk >> 5) * 18 + nt, lane);
      acc = WMMA_BF16(a, b, acc);
    }
    int cn = lane & 15, r0 = (lane >> 4) << 3;
    int col = n0 + cn;
    if (col < 192){
      float sc_q = (col < 96) ? 0.17677669529663687f : 1.f;   // q * 32^-0.5
      #pragma unroll
      for (int j = 0; j < 8; ++j)
        qk[(m0 + r0 + j) * QK_LD + col] = (bf16_t)((acc[j] + qkvb[col]) * sc_q);
    } else {
      int hd = col - 192;
      #pragma unroll
      for (int j = 0; j < 8; ++j)                // contiguous per lane (8 tokens)
        vT[hd * VT_LD + (m0 + r0 + j)] = (bf16_t)(acc[j] + qkvb[col]);
    }
  }
  __syncthreads();

  // ---- per-head attention ----
  for (int h = 0; h < 3; ++h){
    // scores = q @ k^T, one K=32 WMMA per 16x16 tile; fused bias + mask
    for (int t = wv; t < 49; t += 8){
      int mt = t / 7, nt = t - mt * 7;
      int m0 = mt * 16, n0 = nt * 16;
      f32x8 acc = {};
      bf16x16 a = ldA(qk + m0 * QK_LD + h * 32,      QK_LD, lane);  // q rows
      bf16x16 b = ldA(qk + n0 * QK_LD + 96 + h * 32, QK_LD, lane);  // k^T (A-style)
      acc = WMMA_BF16(a, b, acc);
      int cn = lane & 15, r0 = (lane >> 4) << 3;
      #pragma unroll
      for (int j = 0; j < 8; ++j){
        int r = m0 + r0 + j, c = n0 + cn;
        float v;
        if (r < 98){
          if (c < 98){
            int rd = r / 49, rrm = r - rd * 49, rh = rrm / 7, rw = rrm - rh * 7;
            int cd = c / 49, crm = c - cd * 49, ch = crm / 7, cw = crm - ch * 7;
            int idx = (rd - cd + 1) * 169 + (rh - ch + 6) * 13 + (rw - cw + 6);
            v = acc[j] + rpbs[idx * 3 + h] + msk[r * 98 + c];
          } else {
            v = -1e30f;               // padded key -> softmax weight 0
          }
        } else {
          v = 0.f;                    // padded query row -> zero prob row
        }
        sc[r * SC_LD + c] = v;
      }
    }
    __syncthreads();

    // softmax over keys, wave per row; zero-fill pad cols 112..127
    for (int r = wv; r < 98; r += 8){
      float vv[4]; float m = -1e30f;
      #pragma unroll
      for (int i = 0; i < 4; ++i){
        int c = lane + 32 * i;
        vv[i] = (c < 112) ? sc[r * SC_LD + c] : -1e30f;
        m = fmaxf(m, vv[i]);
      }
      m = wmaxr(m);
      float ee[4]; float s = 0.f;
      #pragma unroll
      for (int i = 0; i < 4; ++i){
        int c = lane + 32 * i;
        ee[i] = (c < 112) ? __expf(vv[i] - m) : 0.f;
        s += ee[i];
      }
      s = wsum(s);
      float inv = 1.f / s;
      #pragma unroll
      for (int i = 0; i < 4; ++i){
        int c = lane + 32 * i;
        sc[r * SC_LD + c] = (c < 112) ? ee[i] * inv : 0.f;
      }
    }
    __syncthreads();

    // out_h = probs(112x128) @ v(128x32): B from v^T via contiguous A-style load
    for (int t = wv; t < 14; t += 8){
      int mt = t >> 1, n0 = (t & 1) << 4;
      int m0 = mt * 16;
      f32x8 acc = {};
      #pragma unroll
      for (int kk = 0; kk < 128; kk += 32){
        bf16x16 a = ldAf(sc + m0 * SC_LD + kk, SC_LD, lane);
        bf16x16 b = ldA(vT + (h * 32 + n0) * VT_LD + kk, VT_LD, lane);
        acc = WMMA_BF16(a, b, acc);
      }
      int cn = lane & 15, r0 = (lane >> 4) << 3;
      #pragma unroll
      for (int j = 0; j < 8; ++j)
        ao[(m0 + r0 + j) * AO_LD + h * 32 + n0 + cn] = (bf16_t)acc[j];
    }
    __syncthreads();
  }

  // re-warm the scattered residual rows of x (evicted during attention)
  for (int i = threadIdx.x; i < 98 * 6; i += 256){
    int t = i / 6;
    __builtin_prefetch((const char*)(x + (size_t)tokpos[t] * 96) + (i % 6) * 64, 0, 3);
  }

  // ---- proj + first residual, scatter back (window reverse + unshift) ----
  for (int t = wv; t < 42; t += 8){
    int mt = t / 6, nt = t - mt * 6;
    int m0 = mt * 16, n0 = nt * 16;
    f32x8 acc = {};
    #pragma unroll
    for (int kk = 0; kk < 96; kk += 32){
      bf16x16 a = ldA(ao + m0 * AO_LD + kk, AO_LD, lane);
      bf16x16 b = ldBsw(wproj, (kk >> 5) * 6 + nt, lane);
      acc = WMMA_BF16(a, b, acc);
    }
    int cn = lane & 15, r0 = (lane >> 4) << 3;
    #pragma unroll
    for (int j = 0; j < 8; ++j){
      int r = m0 + r0 + j;
      if (r < 98){
        int col = n0 + cn;
        size_t gp = (size_t)tokpos[r] * 96 + col;
        xres[gp] = x[gp] + acc[j] + projb[col];
      }
    }
  }
}

// ---------------------------------------------------------------------------
// Kernel B: LN2 + fc1 + exact GELU + fc2 + second residual, 64 tokens/block.
// ---------------------------------------------------------------------------
__global__ __launch_bounds__(256) void mlp_kernel(
    const float* __restrict__ xres,
    const float* __restrict__ n2w, const float* __restrict__ n2b,
    const bf16_t* __restrict__ wfc1, const float* __restrict__ fc1b,
    const bf16_t* __restrict__ wfc2, const float* __restrict__ fc2b,
    float* __restrict__ out)
{
  __shared__ bf16_t xn[64 * XN_LD];    // 13312 B
  __shared__ bf16_t hid[64 * HID_LD];  // 50176 B -> 63488 total
  const int lane = threadIdx.x & 31;
  const int wv   = threadIdx.x >> 5;
  const size_t tt0 = (size_t)blockIdx.x * 64;

  for (int it = 0; it < 8; ++it){
    int r = wv + it * 8;
    const float* xp = xres + (tt0 + r) * 96;
    float v0 = xp[lane], v1 = xp[lane + 32], v2 = xp[lane + 64];
    float s  = wsum(v0 + v1 + v2);
    float sq = wsum(v0*v0 + v1*v1 + v2*v2);
    float mean = s * (1.f / 96.f);
    float rstd = rsqrtf(sq * (1.f / 96.f) - mean * mean + 1e-5f);
    xn[r*XN_LD + lane     ] = (bf16_t)((v0 - mean) * rstd * n2w[lane     ] + n2b[lane     ]);
    xn[r*XN_LD + lane + 32] = (bf16_t)((v1 - mean) * rstd * n2w[lane + 32] + n2b[lane + 32]);
    xn[r*XN_LD + lane + 64] = (bf16_t)((v2 - mean) * rstd * n2w[lane + 64] + n2b[lane + 64]);
  }
  __syncthreads();

  // fc1 (96 -> 384) + exact GELU
  for (int t = wv; t < 96; t += 8){
    int mt = t / 24, nt = t - mt * 24;
    int m0 = mt * 16, n0 = nt * 16;
    f32x8 acc = {};
    #pragma unroll
    for (int kk = 0; kk < 96; kk += 32){
      bf16x16 a = ldA(xn + m0 * XN_LD + kk, XN_LD, lane);
      bf16x16 b = ldBsw(wfc1, (kk >> 5) * 24 + nt, lane);
      acc = WMMA_BF16(a, b, acc);
    }
    int cn = lane & 15, r0 = (lane >> 4) << 3;
    #pragma unroll
    for (int j = 0; j < 8; ++j){
      int col = n0 + cn;
      float g = acc[j] + fc1b[col];
      g = 0.5f * g * (1.f + erff(g * 0.70710678118654752f));
      hid[(m0 + r0 + j) * HID_LD + col] = (bf16_t)g;
    }
  }
  __syncthreads();

  // fc2 (384 -> 96) + second residual
  for (int t = wv; t < 24; t += 8){
    int mt = t / 6, nt = t - mt * 6;
    int m0 = mt * 16, n0 = nt * 16;
    f32x8 acc = {};
    #pragma unroll
    for (int kk = 0; kk < 384; kk += 32){
      bf16x16 a = ldA(hid + m0 * HID_LD + kk, HID_LD, lane);
      bf16x16 b = ldBsw(wfc2, (kk >> 5) * 6 + nt, lane);
      acc = WMMA_BF16(a, b, acc);
    }
    int cn = lane & 15, r0 = (lane >> 4) << 3;
    #pragma unroll
    for (int j = 0; j < 8; ++j){
      int col = n0 + cn;
      size_t gi = (tt0 + m0 + r0 + j) * 96 + col;
      out[gi] = xres[gi] + acc[j] + fc2b[col];
    }
  }
}

extern "C" void kernel_launch(void* const* d_in, const int* in_sizes, int n_in,
                              void* d_out, int out_size, void* d_ws, size_t ws_size,
                              hipStream_t stream)
{
  (void)in_sizes; (void)n_in; (void)out_size; (void)ws_size;
  const float* x     = (const float*)d_in[0];
  const float* mask  = (const float*)d_in[1];
  const float* n1w   = (const float*)d_in[2];
  const float* n1b   = (const float*)d_in[3];
  const float* qkvw  = (const float*)d_in[4];
  const float* qkvb  = (const float*)d_in[5];
  const float* rpb   = (const float*)d_in[6];
  const float* projw = (const float*)d_in[7];
  const float* projb = (const float*)d_in[8];
  const float* n2w   = (const float*)d_in[9];
  const float* n2b   = (const float*)d_in[10];
  const float* fc1w  = (const float*)d_in[11];
  const float* fc1b  = (const float*)d_in[12];
  const float* fc2w  = (const float*)d_in[13];
  const float* fc2b  = (const float*)d_in[14];

  char* ws = (char*)d_ws;
  bf16_t* wqkv  = (bf16_t*)(ws + 0);        // 96*288  bf16 (fragment-swizzled)
  bf16_t* wproj = (bf16_t*)(ws + 55296);    // 96*96   bf16
  bf16_t* wfc1  = (bf16_t*)(ws + 73728);    // 96*384  bf16
  bf16_t* wfc2  = (bf16_t*)(ws + 147456);   // 384*96  bf16
  float*  xres  = (float*)(ws + 221184);    // 401408*96 f32

  swz_bf16<<<(27648 + 255) / 256, 256, 0, stream>>>(qkvw, wqkv, 288, 18, 27648);
  swz_bf16<<<(9216  + 255) / 256, 256, 0, stream>>>(projw, wproj, 96, 6, 9216);
  swz_bf16<<<(36864 + 255) / 256, 256, 0, stream>>>(fc1w, wfc1, 384, 24, 36864);
  swz_bf16<<<(36864 + 255) / 256, 256, 0, stream>>>(fc2w, wfc2, 96, 6, 36864);

  const int ATTN_LDS = 221600;  // < 320 KB CDNA5 WGP LDS
  hipFuncSetAttribute(reinterpret_cast<const void*>(attn_kernel),
                      hipFuncAttributeMaxDynamicSharedMemorySize, ATTN_LDS);

  attn_kernel<<<4096, 256, ATTN_LDS, stream>>>(x, mask, n1w, n1b, qkvb, rpb, projb,
                                               wqkv, wproj, xres);
  mlp_kernel<<<6272, 256, 0, stream>>>(xres, n2w, n2b, wfc1, fc1b, wfc2, fc2b,
                                       (float*)d_out);
}
